// CANN_Simple_18854906429894
// MI455X (gfx1250) — matrix operand order repaired and checked by reading, not compiled, not run on hardware
//
#include <hip/hip_runtime.h>

typedef __attribute__((ext_vector_type(16))) _Float16 v16h;
typedef __attribute__((ext_vector_type(8)))  float    v8f;

#define NEG_SLOPE 0.2f
#define ORD_NEG_INF 0x007FFFFFu   // f2ord(-inf)

__device__ __forceinline__ unsigned f2ord(float f) {
    unsigned b = __float_as_uint(f);
    return b ^ (unsigned)((((int)b) >> 31) | 0x80000000);
}
__device__ __forceinline__ float ord2f(unsigned o) {
    unsigned b = (o & 0x80000000u) ? (o ^ 0x80000000u) : ~o;
    return __uint_as_float(b);
}
__device__ __forceinline__ float lrelu(float v) { return v > 0.f ? v : NEG_SLOPE * v; }
__device__ __forceinline__ bool finitef(float f) {
    return ((__float_as_uint(f) >> 23) & 0xFF) != 0xFF;
}

// ---------------- fills / conversions ----------------
__global__ void kFillF32(float* p, int n, float v) {
    int i = blockIdx.x * blockDim.x + threadIdx.x;
    if (i < n) p[i] = v;
}
__global__ void kFillU32(unsigned* p, int n, unsigned v) {
    int i = blockIdx.x * blockDim.x + threadIdx.x;
    if (i < n) p[i] = v;
}
// Build padded f16 A = [x | K | 0-pad], rows padded to Npad, cols to lda
__global__ void kBuildXaug(const float* __restrict__ x, const float* __restrict__ Karr,
                           _Float16* __restrict__ A, int N, int Npad, int IN_, int lda) {
    int t = blockIdx.x * blockDim.x + threadIdx.x;
    int tot = Npad * lda;
    if (t >= tot) return;
    int n = t / lda, c = t - n * lda;
    float v = 0.f;
    if (n < N) {
        if (c < IN_) v = x[(size_t)n * IN_ + c];
        else if (c == IN_) v = Karr[n];
    }
    A[t] = (_Float16)v;
}
// Convert row-major W [rows x cols] to f16, zero-padding rows up to prows
__global__ void kConvertW(const float* __restrict__ W, _Float16* __restrict__ Wh,
                          int rows, int cols, int prows) {
    int t = blockIdx.x * blockDim.x + threadIdx.x;
    int tot = prows * cols;
    if (t >= tot) return;
    int r = t / cols;
    Wh[t] = (_Float16)(r < rows ? W[t] : 0.f);
}

// ---------------- WMMA GEMM: C[f32] = A[f16] * B[f16] ----------------
// Persistent-column waves: each wave owns one 16-wide column tile, keeps ALL
// K-step B fragments in registers, and iterates over a strip of M tiles.
// LDA/LDB/LDC are compile-time so all hot-loop addressing folds to immediate
// offsets (A: 2 x b128/k-step, C: 8 stores off one base, B: 16 loads off one base).
template <int KSTEPS, int LDA, int LDB, int LDC>
__global__ void kWmmaGemmT(const _Float16* __restrict__ A,
                           const _Float16* __restrict__ B,
                           float* __restrict__ C,
                           int mtiles, int ntiles, int strip) {
    int wave = (blockIdx.x * blockDim.x + threadIdx.x) >> 5;
    int lane = threadIdx.x & 31;
    int nstrips = (mtiles + strip - 1) / strip;
    if (wave >= nstrips * ntiles) return;          // wave-uniform exit (EXEC stays full)
    int tn  = wave % ntiles;
    int tm0 = (wave / ntiles) * strip;
    int half = lane >> 4;        // selects K sub-range for A/B fragments
    int mrow = lane & 15;        // A: row in tile; B/C: column in tile

    // Load all B fragments once (reused across the whole M strip).
    // B 32x16 layout: lanes 0-15 hold K=k0+{0..7,16..23} of col=lane, lanes 16-31 K=k0+{8..15,24..31}.
    v16h breg[KSTEPS];
#pragma unroll
    for (int kk = 0; kk < KSTEPS; ++kk) {
        const _Float16* bcol = B + (size_t)(kk * 32 + half * 8) * LDB + tn * 16 + mrow;
#pragma unroll
        for (int j = 0; j < 16; ++j) breg[kk][j] = bcol[(j < 8 ? j : j + 8) * LDB];
    }

    int tmEnd = tm0 + strip;
    if (tmEnd > mtiles) tmEnd = mtiles;
    for (int tm = tm0; tm < tmEnd; ++tm) {
        const _Float16* arow0 = A + (size_t)(tm * 16 + mrow) * LDA + half * 8;
        if (tm + 1 < tmEnd)
            __builtin_prefetch(arow0 + (size_t)16 * LDA, 0, 1);   // next tile's A line
        v8f c = {};
#pragma unroll
        for (int kk = 0; kk < KSTEPS; ++kk) {
            // A 16x32 layout: half 0 -> K {0..7,16..23}, half 1 -> {8..15,24..31}
            v16h a;
            const _Float16* arow = arow0 + kk * 32;
#pragma unroll
            for (int j = 0; j < 8; ++j) { a[j] = arow[j]; a[8 + j] = arow[16 + j]; }
            c = __builtin_amdgcn_wmma_f32_16x16x32_f16(false, a, false, breg[kk],
                                                       (short)0, c, false, false);
        }
        // C/D layout: VGPR r, lane L -> row = r + 8*(L>=16), col = L&15
        float* cp = C + (size_t)(tm * 16 + half * 8) * LDC + tn * 16 + mrow;
#pragma unroll
        for (int r = 0; r < 8; ++r)
            cp[r * LDC] = c[r];
    }
}

// ---------------- attention scores ----------------
__global__ void kScores(const float* __restrict__ xh, int ld,
                        const float* __restrict__ att_s, const float* __restrict__ att_d,
                        float* __restrict__ as_, float* __restrict__ ad_,
                        int N, int H, int C) {
    int t = blockIdx.x * blockDim.x + threadIdx.x;
    if (t >= N * H) return;
    int n = t / H, h = t - n * H;
    const float* row = xh + (size_t)n * ld + h * C;
    const float* as = att_s + h * C;
    const float* ad = att_d + h * C;
    float ss = 0.f, sd = 0.f;
    for (int c = 0; c < C; ++c) { float v = row[c]; ss += v * as[c]; sd += v * ad[c]; }
    as_[t] = ss; ad_[t] = sd;
}

// ---------------- edge passes ----------------
__global__ void kEdgeMax(const int* __restrict__ ei, int E, int N,
                         const float* __restrict__ as_, const float* __restrict__ ad_,
                         unsigned* __restrict__ mord, int H) {
    int t = blockIdx.x * blockDim.x + threadIdx.x;
    int tot = (E + N) * H;
    if (t >= tot) return;
    int e = t / H, h = t - e * H;
    int s = e < E ? ei[e] : e - E;
    int d = e < E ? ei[E + e] : e - E;
    float v = lrelu(as_[s * H + h] + ad_[d * H + h]);
    atomicMax(&mord[d * H + h], f2ord(v));
}
__global__ void kFinalizeM(unsigned* __restrict__ mord, int n) {
    int t = blockIdx.x * blockDim.x + threadIdx.x;
    if (t >= n) return;
    float f = ord2f(mord[t]);
    if (!finitef(f)) f = 0.f;
    ((float*)mord)[t] = f;
}
__global__ void kEdgeDenom(const int* __restrict__ ei, int E, int N,
                           const float* __restrict__ as_, const float* __restrict__ ad_,
                           const float* __restrict__ m, float* __restrict__ denom, int H) {
    int t = blockIdx.x * blockDim.x + threadIdx.x;
    int tot = (E + N) * H;
    if (t >= tot) return;
    int e = t / H, h = t - e * H;
    int s = e < E ? ei[e] : e - E;
    int d = e < E ? ei[E + e] : e - E;
    float v = lrelu(as_[s * H + h] + ad_[d * H + h]);
    atomicAdd(&denom[d * H + h], __expf(v - m[d * H + h]));
}
__global__ void kEdgeAgg(const int* __restrict__ ei, int E, int N,
                         const float* __restrict__ as_, const float* __restrict__ ad_,
                         const float* __restrict__ m, const float* __restrict__ denom,
                         const float* __restrict__ xh, int ld,
                         float* __restrict__ agg, int H, int C) {
    long long t = (long long)blockIdx.x * blockDim.x + threadIdx.x;
    long long tot = (long long)(E + N) * H * C;
    if (t >= tot) return;
    int c = (int)(t % C);
    long long eh = t / C;
    int h = (int)(eh % H);
    int e = (int)(eh / H);
    int s = e < E ? ei[e] : e - E;
    int d = e < E ? ei[E + e] : e - E;
    float v = lrelu(as_[s * H + h] + ad_[d * H + h]);
    float ex = __expf(v - m[d * H + h]);
    float alpha = ex / (denom[d * H + h] + 1e-16f);
    atomicAdd(&agg[(size_t)d * H * C + h * C + c], alpha * xh[(size_t)s * ld + h * C + c]);
}

// head-mean + bias + relu -> f16 h (zero-padded rows for next GEMM)
__global__ void kCombine1(const float* __restrict__ agg, const float* __restrict__ b1,
                          _Float16* __restrict__ hh, int N, int Npad) {
    int t = blockIdx.x * blockDim.x + threadIdx.x;
    if (t >= Npad * 32) return;
    int n = t / 32, c = t - n * 32;
    float v = 0.f;
    if (n < N) {
        float s = 0.f;
        for (int h = 0; h < 4; ++h) s += agg[(size_t)n * 128 + h * 32 + c];
        v = s * 0.25f + b1[c];
        v = v > 0.f ? v : 0.f;
    }
    hh[t] = (_Float16)v;
}
__global__ void kFinalOut(const float* __restrict__ agg2, const float* __restrict__ b2,
                          float* __restrict__ out, int N) {
    int t = blockIdx.x * blockDim.x + threadIdx.x;
    if (t >= N * 32) return;
    out[t] = agg2[t] + b2[t & 31];
}

static inline int nblk(long long n, int b) { return (int)((n + b - 1) / b); }

extern "C" void kernel_launch(void* const* d_in, const int* in_sizes, int n_in,
                              void* d_out, int out_size, void* d_ws, size_t ws_size,
                              hipStream_t stream) {
    (void)n_in; (void)out_size; (void)ws_size;
    const float* x    = (const float*)d_in[0];
    const float* Karr = (const float*)d_in[1];
    const int*   ei   = (const int*)d_in[2];
    const float* W1   = (const float*)d_in[3];
    const float* as1  = (const float*)d_in[4];
    const float* ad1  = (const float*)d_in[5];
    const float* b1   = (const float*)d_in[6];
    const float* W2   = (const float*)d_in[7];
    const float* as2  = (const float*)d_in[8];
    const float* ad2  = (const float*)d_in[9];
    const float* b2   = (const float*)d_in[10];
    float* out = (float*)d_out;

    const int N   = in_sizes[1];          // 50000
    const int E   = in_sizes[2] / 2;      // 800000
    // Model dims fixed by the reference: IN=128 (+1 for K), H=4, C=32, OUT=32.
    const int IN_ = 128;
    const int H = 4, C = 32;
    const int Npad = ((N + 15) / 16) * 16;
    const int KP1  = 160;                 // 129 padded to K-multiple of 32
    const int F1   = 128;                 // H*C

    char* ws = (char*)d_ws;
    size_t off = 0;
    auto alloc = [&](size_t bytes) -> void* {
        void* p = ws + off;
        off = (off + bytes + 255) & ~(size_t)255;
        return p;
    };
    _Float16* Ah   = (_Float16*)alloc((size_t)Npad * KP1 * 2);
    _Float16* W1h  = (_Float16*)alloc((size_t)KP1 * F1 * 2);
    _Float16* W2h  = (_Float16*)alloc((size_t)32 * 32 * 2);
    float*    xh1  = (float*)alloc((size_t)Npad * F1 * 4);
    float*    as1n = (float*)alloc((size_t)N * H * 4);
    float*    ad1n = (float*)alloc((size_t)N * H * 4);
    unsigned* m1   = (unsigned*)alloc((size_t)N * H * 4);
    float*    den1 = (float*)alloc((size_t)N * H * 4);
    float*    agg1 = (float*)alloc((size_t)N * F1 * 4);
    _Float16* hh   = (_Float16*)alloc((size_t)Npad * 32 * 2);
    float*    xh2  = (float*)alloc((size_t)Npad * 32 * 4);
    float*    as2n = (float*)alloc((size_t)N * 4);
    float*    ad2n = (float*)alloc((size_t)N * 4);
    unsigned* m2   = (unsigned*)alloc((size_t)N * 4);
    float*    den2 = (float*)alloc((size_t)N * 4);
    float*    agg2 = (float*)alloc((size_t)N * 32 * 4);

    const int B = 256;
    const int ET = E + N;   // edges incl. self loops

    // ---- prep / init ----
    kBuildXaug<<<nblk((long long)Npad * KP1, B), B, 0, stream>>>(x, Karr, Ah, N, Npad, IN_, KP1);
    kConvertW<<<nblk((long long)KP1 * F1, B), B, 0, stream>>>(W1, W1h, IN_ + 1, F1, KP1);
    kConvertW<<<nblk(32 * 32, B), B, 0, stream>>>(W2, W2h, 32, 32, 32);
    kFillU32<<<nblk(N * H, B), B, 0, stream>>>(m1, N * H, ORD_NEG_INF);
    kFillF32<<<nblk(N * H, B), B, 0, stream>>>(den1, N * H, 0.f);
    kFillF32<<<nblk((long long)N * F1, B), B, 0, stream>>>(agg1, N * F1, 0.f);
    kFillU32<<<nblk(N, B), B, 0, stream>>>(m2, N, ORD_NEG_INF);
    kFillF32<<<nblk(N, B), B, 0, stream>>>(den2, N, 0.f);
    kFillF32<<<nblk((long long)N * 32, B), B, 0, stream>>>(agg2, N * 32, 0.f);

    // ---- layer 1: xh1 = Xaug_h @ W1_h  (KSTEPS=5, strip-mined WMMA) ----
    {
        int mt = Npad / 16, nt = F1 / 16, strip = 8;
        long long waves = (long long)((mt + strip - 1) / strip) * nt;
        kWmmaGemmT<5, 160, 128, 128><<<nblk(waves * 32, B), B, 0, stream>>>(
            Ah, W1h, xh1, mt, nt, strip);
    }
    kScores<<<nblk(N * H, B), B, 0, stream>>>(xh1, F1, as1, ad1, as1n, ad1n, N, H, C);
    kEdgeMax<<<nblk((long long)ET * H, B), B, 0, stream>>>(ei, E, N, as1n, ad1n, m1, H);
    kFinalizeM<<<nblk(N * H, B), B, 0, stream>>>(m1, N * H);
    kEdgeDenom<<<nblk((long long)ET * H, B), B, 0, stream>>>(ei, E, N, as1n, ad1n,
        (const float*)m1, den1, H);
    kEdgeAgg<<<nblk((long long)ET * H * C, B), B, 0, stream>>>(ei, E, N, as1n, ad1n,
        (const float*)m1, den1, xh1, F1, agg1, H, C);
    kCombine1<<<nblk((long long)Npad * 32, B), B, 0, stream>>>(agg1, b1, hh, N, Npad);

    // ---- layer 2: xh2 = h_h @ W2_h  (KSTEPS=1) ----
    {
        int mt = Npad / 16, nt = 2, strip = 4;
        long long waves = (long long)((mt + strip - 1) / strip) * nt;
        kWmmaGemmT<1, 32, 32, 32><<<nblk(waves * 32, B), B, 0, stream>>>(
            hh, W2h, xh2, mt, nt, strip);
    }
    kScores<<<nblk(N, B), B, 0, stream>>>(xh2, 32, as2, ad2, as2n, ad2n, N, 1, 32);
    kEdgeMax<<<nblk(ET, B), B, 0, stream>>>(ei, E, N, as2n, ad2n, m2, 1);
    kFinalizeM<<<nblk(N, B), B, 0, stream>>>(m2, N);
    kEdgeDenom<<<nblk(ET, B), B, 0, stream>>>(ei, E, N, as2n, ad2n, (const float*)m2, den2, 1);
    kEdgeAgg<<<nblk((long long)ET * 32, B), B, 0, stream>>>(ei, E, N, as2n, ad2n,
        (const float*)m2, den2, xh2, 32, agg2, 1, 32);
    kFinalOut<<<nblk((long long)N * 32, B), B, 0, stream>>>(agg2, b2, out, N);
}
